// MultiHeadAttention_40853728920331
// MI455X (gfx1250) — matrix operand order repaired and checked by reading, not compiled, not run on hardware
//
#include <hip/hip_runtime.h>
#include <hip/hip_bf16.h>

// ---------------------------------------------------------------------------
// MHA (Whisper-style) for MI455X / gfx1250, wave32, WMMA f16->f32.
// Outputs: d_out = [ out (2*2048*1024 f32) | qk (2*16*2048*2048 f32) ]
// qk (537 MB, write-once) is stored with non-temporal hints so it does not
// thrash the 192 MB L2 that caches the Q/K/V working set (24 MB).
// Workspace layout (bytes), total 50,331,648 (48 MiB):
//   xh   @ 0         : f16 [4096][1024]
//   wqT  @ 8388608   : f16 [1024][1024]  (transposed: [n][k])
//   wkT  @ 10485760
//   wvT  @ 12582912
//   woT  @ 14680064
//   qh   @ 16777216  : f16 [4096][1024]  (= (x@Wq+bq) * 64^-0.25)
//   kh   @ 25165824  : f16 [4096][1024]  (= (x@Wk)    * 64^-0.25)
//   vt   @ 33554432  : f16 [2][16][64][2048]  (V stored d-major)
//   wvh  @ 41943040  : f16 [4096][1024]  (softmax(qk) @ V)
// ---------------------------------------------------------------------------

typedef _Float16 f16;
typedef __attribute__((ext_vector_type(16))) _Float16 v16h;
typedef __attribute__((ext_vector_type(8)))  _Float16 v8h;
typedef __attribute__((ext_vector_type(8)))  float    v8f;

#define DEVI __device__ __forceinline__

constexpr int Bc  = 2;
constexpr int Sc  = 2048;
constexpr int NS  = 1024;
constexpr int Hc  = 16;
constexpr int Dc  = 64;
constexpr int Mtot = Bc * Sc;                         // 4096
constexpr float QK_SCALE = 0.35355339059327379f;      // 64^-0.25

union V16U { v16h v; v8h h[2]; };

// A-matrix 16x32 f16 fragment: lane l holds row (l&15);
// lanes hold K = kb..kb+15 contiguous, kb=(l>>4)*16.
DEVI v16h load_a_frag(const f16* base, int ld, int row0, int k0, int lane) {
  const int row = row0 + (lane & 15);
  const int kb  = (lane >> 4) * 16;
  const f16* p = base + (long)row * ld + k0 + kb;
  V16U u;
  u.h[0] = *(const v8h*)(p);        // K = kb .. kb+7
  u.h[1] = *(const v8h*)(p + 8);    // K = kb+8 .. kb+15
  return u.v;
}

// B-matrix 32x16 f16 fragment, from a K-major ("B transposed") array bt[N][K]:
// lane l holds column n0+(l&15); lanes 0-15 hold K=0..15, lanes 16-31 K=16..31.
DEVI v16h load_bt_frag(const f16* bt, int ld, int n0, int k0, int lane) {
  const int n  = n0 + (lane & 15);
  const int kb = (lane >> 4) * 16;
  const f16* p = bt + (long)n * ld + k0 + kb;
  V16U u;
  u.h[0] = *(const v8h*)(p);
  u.h[1] = *(const v8h*)(p + 8);
  return u.v;
}

DEVI v8f wmma16(v16h a, v16h b, v8f c) {
  // D = A(16x32) * B(32x16) + C ; f32 accumulate
  return __builtin_amdgcn_wmma_f32_16x16x32_f16(false, a, false, b,
                                                (short)0, c, false, false);
}

// -------------------------- conversion kernels -----------------------------

__global__ void k_cvt_x(const float* __restrict__ x, f16* __restrict__ xh, int n) {
  int i = blockIdx.x * blockDim.x + threadIdx.x;
  const int stride = gridDim.x * blockDim.x;
  for (; i < n; i += stride) xh[i] = (f16)x[i];
}

// W[k][n] (f32) -> wT[n][k] (f16); blockIdx.z selects which weight.
__global__ void k_cvt_wT(const float* __restrict__ Wq, const float* __restrict__ Wk,
                         const float* __restrict__ Wv, const float* __restrict__ Wo,
                         f16* __restrict__ wqT, f16* __restrict__ wkT,
                         f16* __restrict__ wvT, f16* __restrict__ woT) {
  const int idx = blockIdx.x * blockDim.x + threadIdx.x;   // 0 .. 1024*1024-1
  const int k = idx >> 10;
  const int n = idx & 1023;
  const float* W;
  f16* T;
  switch (blockIdx.z) {
    case 0:  W = Wq; T = wqT; break;
    case 1:  W = Wk; T = wkT; break;
    case 2:  W = Wv; T = wvT; break;
    default: W = Wo; T = woT; break;
  }
  T[(long)n * NS + k] = (f16)W[(long)k * NS + n];   // coalesced read
}

// ------------------------------ QKV GEMM -----------------------------------
// C[4096,1024] = xh @ W(+bias), one wave per 16x16 tile. z: 0=q,1=k,2=v.

__global__ void k_qkv(const f16* __restrict__ xh,
                      const f16* __restrict__ wqT, const f16* __restrict__ wkT,
                      const f16* __restrict__ wvT,
                      const float* __restrict__ bq, const float* __restrict__ bv,
                      f16* __restrict__ qh, f16* __restrict__ kh,
                      f16* __restrict__ vt) {
  const int lane  = threadIdx.x & 31;
  const int m0    = blockIdx.x * 16;
  const int n0    = blockIdx.y * 16;
  const int which = blockIdx.z;
  const f16* bt = (which == 0) ? wqT : ((which == 1) ? wkT : wvT);

  v8f c = {};
  for (int k0 = 0; k0 < NS; k0 += 32) {
    if (k0 + 32 < NS)
      __builtin_prefetch(bt + (long)(n0 + (lane & 15)) * NS + k0 + 32, 0, 0);
    v16h a = load_a_frag(xh, NS, m0, k0, lane);
    v16h b = load_bt_frag(bt, NS, n0, k0, lane);
    c = wmma16(a, b, c);
  }

  const int n = n0 + (lane & 15);
  const int rowHi = (lane >> 4) * 8;
  const float bias  = (which == 0) ? bq[n] : ((which == 2) ? bv[n] : 0.0f);
  const float scale = (which == 2) ? 1.0f : QK_SCALE;
#pragma unroll
  for (int j = 0; j < 8; ++j) {
    const int m = m0 + j + rowHi;                 // token index (C layout row)
    const float val = (c[j] + bias) * scale;
    if (which == 0) {
      qh[(long)m * NS + n] = (f16)val;
    } else if (which == 1) {
      kh[(long)m * NS + n] = (f16)val;
    } else {
      const int b = m >> 11, s = m & 2047;
      const int h = n >> 6,  d = n & 63;
      vt[(((long)(b * Hc + h) * Dc + d) << 11) + s] = (f16)val;  // d-major
    }
  }
}

// --------------------------- fused flash attention -------------------------
// One wave handles 16 query rows of one (b,h). Streams qk scores to d_out
// with non-temporal stores, keeps online softmax + P@V accumulation on-chip.

__global__ void k_attn(const f16* __restrict__ qh, const f16* __restrict__ kh,
                       const f16* __restrict__ vt,
                       float* __restrict__ qk_out, f16* __restrict__ wvh) {
  __shared__ __align__(32) f16 lds_p[16 * 32];    // P tile staging (1 KiB)

  const int lane = threadIdx.x & 31;
  const int q0 = blockIdx.x * 16;
  const int h  = blockIdx.y;
  const int b  = blockIdx.z;
  const int bh = b * Hc + h;
  const long qk_base = (long)bh * Sc * Sc;
  const int rowHi = (lane >> 4) * 8;              // C-layout row offset
  const int col   = lane & 15;

  const f16* qb = qh + (long)b * Sc * NS + h * Dc;   // [s][d], ld = NS
  const f16* kb = kh + (long)b * Sc * NS + h * Dc;   // [s][d], ld = NS
  const f16* vb = vt + (long)bh * Dc * Sc;           // [d][s], ld = Sc

  // Q fragments for the whole d=64 range (K split as 0..31, 32..63)
  const v16h qa0 = load_a_frag(qb, NS, q0, 0, lane);
  const v16h qa1 = load_a_frag(qb, NS, q0, 32, lane);

  v8f o0 = {}, o1 = {}, o2 = {}, o3 = {};
  float Mrun[8], Lrun[8];
#pragma unroll
  for (int j = 0; j < 8; ++j) { Mrun[j] = -3.0e38f; Lrun[j] = 0.0f; }

  const int qlast = q0 + 15;
  for (int kblk = 0; kblk < Sc / 32; ++kblk) {
    const int kstart = kblk * 32;

    if (kstart > qlast) {          // fully above the diagonal: qk := -1e9
#pragma unroll
      for (int j = 0; j < 8; ++j) {
        float* p = qk_out + qk_base + (long)(q0 + j + rowHi) * Sc + kstart + col;
        __builtin_nontemporal_store(-1e9f, p);
        __builtin_nontemporal_store(-1e9f, p + 16);
      }
      continue;
    }

    if (kstart + 32 < Sc)
      __builtin_prefetch(kb + (long)(kstart + 32 + col) * NS, 0, 0);

    // ---- scores S[16 x 32] = Qtile @ K^T (4 WMMAs) ----
    v8f s0 = {}, s1 = {};
    {
      v16h kf0 = load_bt_frag(kb, NS, kstart,      0,  lane);
      v16h kf1 = load_bt_frag(kb, NS, kstart,      32, lane);
      s0 = wmma16(qa0, kf0, s0);
      s0 = wmma16(qa1, kf1, s0);
      v16h kf2 = load_bt_frag(kb, NS, kstart + 16, 0,  lane);
      v16h kf3 = load_bt_frag(kb, NS, kstart + 16, 32, lane);
      s1 = wmma16(qa0, kf2, s1);
      s1 = wmma16(qa1, kf3, s1);
    }

    // ---- causal mask, stream qk to HBM (NT), tile row-max ----
    float tmax[8];
#pragma unroll
    for (int j = 0; j < 8; ++j) {
      const int qi  = q0 + j + rowHi;
      const int ki0 = kstart + col;
      const int ki1 = kstart + 16 + col;
      const float v0 = (ki0 <= qi) ? s0[j] : -1e9f;
      const float v1 = (ki1 <= qi) ? s1[j] : -1e9f;
      float* r = qk_out + qk_base + (long)qi * Sc;
      __builtin_nontemporal_store(v0, r + ki0);
      __builtin_nontemporal_store(v1, r + ki1);
      s0[j] = v0; s1[j] = v1;
      tmax[j] = fmaxf(v0, v1);
    }
#pragma unroll
    for (int j = 0; j < 8; ++j) {   // reduce over the 16 lanes holding row j
      float t = tmax[j];
      t = fmaxf(t, __shfl_xor(t, 1, 32));
      t = fmaxf(t, __shfl_xor(t, 2, 32));
      t = fmaxf(t, __shfl_xor(t, 4, 32));
      t = fmaxf(t, __shfl_xor(t, 8, 32));
      tmax[j] = t;
    }

    // ---- online softmax update; stage P into LDS (row-major 16x32) ----
    float alpha[8], rowsum[8];
#pragma unroll
    for (int j = 0; j < 8; ++j) {
      const float newM = fmaxf(Mrun[j], tmax[j]);
      alpha[j] = __expf(Mrun[j] - newM);
      const float p0 = __expf(s0[j] - newM);   // masked (-1e9) underflows to 0
      const float p1 = __expf(s1[j] - newM);
      Mrun[j] = newM;
      o0[j] *= alpha[j]; o1[j] *= alpha[j];
      o2[j] *= alpha[j]; o3[j] *= alpha[j];
      const int row = j + rowHi;
      lds_p[row * 32 + col]      = (f16)p0;
      lds_p[row * 32 + 16 + col] = (f16)p1;
      rowsum[j] = p0 + p1;
    }
#pragma unroll
    for (int j = 0; j < 8; ++j) {
      float t = rowsum[j];
      t += __shfl_xor(t, 1, 32);
      t += __shfl_xor(t, 2, 32);
      t += __shfl_xor(t, 4, 32);
      t += __shfl_xor(t, 8, 32);
      Lrun[j] = Lrun[j] * alpha[j] + t;
    }

    // all 32 lanes of this wave must see the LDS P tile
    asm volatile("s_wait_dscnt 0" ::: "memory");

    // ---- O += P(16x32) @ V(32x64)  (4 WMMAs) ----
    const v16h pa = load_a_frag(lds_p, 32, 0, 0, lane);
    v16h vf0 = load_bt_frag(vb, Sc, 0,  kstart, lane);
    v16h vf1 = load_bt_frag(vb, Sc, 16, kstart, lane);
    v16h vf2 = load_bt_frag(vb, Sc, 32, kstart, lane);
    v16h vf3 = load_bt_frag(vb, Sc, 48, kstart, lane);
    o0 = wmma16(pa, vf0, o0);
    o1 = wmma16(pa, vf1, o1);
    o2 = wmma16(pa, vf2, o2);
    o3 = wmma16(pa, vf3, o3);
  }

  // ---- normalize and store wv tile (f16, [s][h*64+d]) ----
#pragma unroll
  for (int j = 0; j < 8; ++j) {
    const float inv = 1.0f / Lrun[j];           // diagonal guarantees L > 0
    const int qi = q0 + j + rowHi;
    f16* dst = wvh + ((long)b * Sc + qi) * NS + h * Dc;
    dst[ 0 + col] = (f16)(o0[j] * inv);
    dst[16 + col] = (f16)(o1[j] * inv);
    dst[32 + col] = (f16)(o2[j] * inv);
    dst[48 + col] = (f16)(o3[j] * inv);
  }
}

// --------------------------- output projection -----------------------------

__global__ void k_oproj(const f16* __restrict__ wvh, const f16* __restrict__ woT,
                        const float* __restrict__ bo, float* __restrict__ out) {
  const int lane = threadIdx.x & 31;
  const int m0 = blockIdx.x * 16;
  const int n0 = blockIdx.y * 16;

  v8f c = {};
  for (int k0 = 0; k0 < NS; k0 += 32) {
    if (k0 + 32 < NS)
      __builtin_prefetch(woT + (long)(n0 + (lane & 15)) * NS + k0 + 32, 0, 0);
    v16h a = load_a_frag(wvh, NS, m0, k0, lane);
    v16h b = load_bt_frag(woT, NS, n0, k0, lane);
    c = wmma16(a, b, c);
  }

  const int n = n0 + (lane & 15);
  const int rowHi = (lane >> 4) * 8;
  const float bias = bo[n];
#pragma unroll
  for (int j = 0; j < 8; ++j) {
    const int m = m0 + j + rowHi;
    __builtin_nontemporal_store(c[j] + bias, out + (long)m * NS + n);
  }
}

// ------------------------------- launcher ----------------------------------

extern "C" void kernel_launch(void* const* d_in, const int* in_sizes, int n_in,
                              void* d_out, int out_size, void* d_ws, size_t ws_size,
                              hipStream_t stream) {
  (void)in_sizes; (void)n_in; (void)out_size; (void)ws_size;

  const float* x  = (const float*)d_in[0];
  // d_in[1] = mask (unused; causal mask synthesized in-kernel)
  const float* Wq = (const float*)d_in[2];
  const float* bq = (const float*)d_in[3];
  const float* Wk = (const float*)d_in[4];
  const float* Wv = (const float*)d_in[5];
  const float* bv = (const float*)d_in[6];
  const float* Wo = (const float*)d_in[7];
  const float* bo = (const float*)d_in[8];

  char* ws = (char*)d_ws;
  f16* xh  = (f16*)(ws + 0);
  f16* wqT = (f16*)(ws + 8388608);
  f16* wkT = (f16*)(ws + 10485760);
  f16* wvT = (f16*)(ws + 12582912);
  f16* woT = (f16*)(ws + 14680064);
  f16* qh  = (f16*)(ws + 16777216);
  f16* kh  = (f16*)(ws + 25165824);
  f16* vt  = (f16*)(ws + 33554432);
  f16* wvh = (f16*)(ws + 41943040);

  float* out_p = (float*)d_out;                       // [2,2048,1024]
  float* qk_p  = out_p + (long)Bc * Sc * NS;          // [2,16,2048,2048]

  // 1. convert inputs to f16 (weights transposed to [n][k])
  k_cvt_x<<<2048, 256, 0, stream>>>(x, xh, Mtot * NS);
  k_cvt_wT<<<dim3(4096, 1, 4), 256, 0, stream>>>(Wq, Wk, Wv, Wo,
                                                 wqT, wkT, wvT, woT);
  // 2. fused QKV projection (scale baked into q,k; v stored d-major)
  k_qkv<<<dim3(Mtot / 16, NS / 16, 3), 32, 0, stream>>>(
      xh, wqT, wkT, wvT, bq, bv, qh, kh, vt);
  // 3. flash attention: streams qk scores (NT), produces wv
  k_attn<<<dim3(Sc / 16, Hc, Bc), 32, 0, stream>>>(qh, kh, vt, qk_p, wvh);
  // 4. output projection
  k_oproj<<<dim3(Mtot / 16, NS / 16), 32, 0, stream>>>(wvh, woT, bo, out_p);
}
